// Attention_50105088475686
// MI455X (gfx1250) — matrix operand order repaired and checked by reading, not compiled, not run on hardware
//
#include <hip/hip_runtime.h>

typedef __attribute__((ext_vector_type(16))) _Float16 v16h;
typedef __attribute__((ext_vector_type(8)))  float    v8f;

#define B_   8
#define N_   1025
#define C_   768
#define H_   12
#define D_   64
#define NP_  1056            // padded kv length = 33*32
#define M_   (B_ * N_)       // 8200 rows for the two big GEMMs
#define SCALE_ 0.125f        // D^-0.5

// ---------------------------------------------------------------- CDNA5 async helpers

// Per-lane async copy of 16 bytes: LDS[ldsoff] = MEM[gsrc] (tracked by ASYNCcnt).
static __device__ inline void async_ld_b128(unsigned ldsoff, const void* gsrc) {
  asm volatile("global_load_async_to_lds_b128 %0, %1, off"
               :: "v"(ldsoff), "v"(gsrc) : "memory");
}
static __device__ inline void wait_async0() {
  asm volatile("s_wait_asynccnt 0" ::: "memory");
}
static __device__ inline unsigned lds_off(const void* p) {
  // generic pointers to LDS carry the byte offset in the low 32 bits
  return (unsigned)(unsigned long long)p;
}

// ---------------------------------------------------------------- WMMA helpers

static __device__ inline v8f wmma_f16(v16h a, v16h b, v8f c) {
  // D = A(16x32 f16) * B(32x16 f16) + C(16x16 f32)
  return __builtin_amdgcn_wmma_f32_16x16x32_f16(
      /*neg_a=*/false, a, /*neg_b=*/false, b,
      /*c_mod=*/(short)0, c, /*reuse_a=*/false, /*reuse_b=*/false);
}

// A fragment (16x32 f16). rowPtr already points at this lane's row (lane&15).
// Layout: lanes 0-15 hold K {0..15}, lanes 16-31 hold K {16..31}
// (contiguous, matching the B fragment loaders).
static __device__ inline v16h load_a_frag(const _Float16* rowPtr, int kbase, int lane) {
  const int g = (lane >> 4) & 1;
  const _Float16* p0 = rowPtr + kbase + g * 16;
  const _Float16* p1 = rowPtr + kbase + g * 16 + 8;
  v16h a;
#pragma unroll
  for (int i = 0; i < 8; ++i) { a[i] = p0[i]; a[8 + i] = p1[i]; }
  return a;
}

// B fragment where the source is stored [col][k] row-major (i.e. B = W^T).
// Lane (&15) = output column, g selects K half: 16 contiguous halves per lane.
static __device__ inline v16h load_bt_frag(const _Float16* w, int ld, int colbase,
                                           int kbase, int lane) {
  const int g = (lane >> 4) & 1;
  const _Float16* p = w + (size_t)(colbase + (lane & 15)) * ld + kbase + g * 16;
  v16h b;
#pragma unroll
  for (int i = 0; i < 16; ++i) b[i] = p[i];
  return b;
}

// B fragment where the source is stored [k][col] row-major (used for V).
static __device__ inline v16h load_b_frag(const _Float16* m, int ld, int kbase,
                                          int colbase, int lane) {
  const int g = (lane >> 4) & 1;
  const _Float16* p = m + (size_t)(kbase + g * 16) * ld + colbase + (lane & 15);
  v16h b;
#pragma unroll
  for (int i = 0; i < 16; ++i) b[i] = p[(size_t)i * ld];
  return b;
}

// ---------------------------------------------------------------- convert

__global__ void f32_to_f16_kernel(const float* __restrict__ in,
                                  _Float16* __restrict__ out, int n) {
  int i = blockIdx.x * blockDim.x + threadIdx.x;
  if (i < n) out[i] = (_Float16)in[i];
}

// ---------------------------------------------------------------- QKV GEMM
// Block = 8 waves = 128 rows x 64 cols; weight chunk (64 cols x 32 k = 4KB)
// shared via async-to-LDS double buffer. Output scattered into padded
// [B,H,NP_,D] f16 q/k/v buffers.

__global__ __launch_bounds__(256) void qkv_gemm_kernel(
    const _Float16* __restrict__ xh, const _Float16* __restrict__ wh,
    _Float16* __restrict__ qb, _Float16* __restrict__ kb, _Float16* __restrict__ vb) {
  __shared__ _Float16 btile[2][64 * 32];
  const int tid = threadIdx.x;
  const int wave = tid >> 5;
  const int lane = tid & 31;
  const int g = (lane >> 4) & 1;
  const int NTC = (3 * C_) / 64;              // 36 column groups of 64
  const int mb = blockIdx.x / NTC;            // 0..64 (row blocks of 128)
  const int nt = blockIdx.x - mb * NTC;
  const int colbase = nt * 64;

  const int arow = mb * 128 + wave * 16 + (lane & 15);
  const _Float16* ap = xh + (size_t)(arow < M_ ? arow : (M_ - 1)) * C_;

  // staging: thread tid copies 8 halves of weight row (colbase + tid/4)
  const int scol = tid >> 2, sseg = tid & 3;
  const _Float16* wsrc = wh + (size_t)(colbase + scol) * C_ + sseg * 8;
  const unsigned ldst0 = lds_off(&btile[0][scol * 32 + sseg * 8]);
  const unsigned ldst1 = lds_off(&btile[1][scol * 32 + sseg * 8]);

  async_ld_b128(ldst0, wsrc);                 // preload chunk 0
  v8f acc[4] = {};
  for (int c = 0; c < C_ / 32; ++c) {         // 24 chunks
    wait_async0();
    __syncthreads();                          // chunk c resident for all waves
    if (c + 1 < C_ / 32)
      async_ld_b128((c & 1) ? ldst0 : ldst1, wsrc + (c + 1) * 32);
    v16h a = load_a_frag(ap, c * 32, lane);
    const _Float16* bt = &btile[c & 1][0];
#pragma unroll
    for (int t = 0; t < 4; ++t)
      acc[t] = wmma_f16(a, load_bt_frag(bt, 32, t * 16, 0, lane), acc[t]);
  }

#pragma unroll
  for (int t = 0; t < 4; ++t) {
    const int col  = colbase + t * 16 + (lane & 15);
    const int msel = col / C_;                 // 0=q 1=k 2=v
    const int rem  = col - msel * C_;
    const int h    = rem >> 6;
    const int d    = rem & 63;
    _Float16* dst = (msel == 0) ? qb : (msel == 1) ? kb : vb;
#pragma unroll
    for (int r = 0; r < 8; ++r) {
      const int orow = mb * 128 + wave * 16 + r + 8 * g;
      if (orow < M_) {
        const int b = orow / N_;
        const int n = orow - b * N_;
        dst[((size_t)(b * H_ + h) * NP_ + n) * D_ + d] = (_Float16)acc[t][r];
      }
    }
  }
}

// ---------------------------------------------------------------- flash attention
// Block = 4 waves sharing one (b,h), covering 4 adjacent 16-row q tiles.
// K and V streamed in 32-row chunks through async double-buffered LDS;
// online softmax; P transposed through per-wave LDS into an A-fragment.

__global__ __launch_bounds__(128) void attn_kernel(
    const _Float16* __restrict__ q, const _Float16* __restrict__ k,
    const _Float16* __restrict__ v, _Float16* __restrict__ ao) {
  __shared__ _Float16 kbuf[2][32 * 64];
  __shared__ _Float16 vbuf[2][32 * 64];
  __shared__ _Float16 p_lds[4][16 * 32];
  const int tid = threadIdx.x;
  const int wave = tid >> 5;
  const int lane = tid & 31;
  const int g = (lane >> 4) & 1;
  const int QB = 17;                           // ceil(65 q-tiles / 4 waves)
  const int bh = blockIdx.x / QB;
  const int qt4 = blockIdx.x - bh * QB;
  const int tq = qt4 * 4 + wave;               // 0..67 (65+ are dummy, guarded)
  const int bb = bh / H_;
  const int hh = bh - bb * H_;

  const _Float16* qp = q + (size_t)bh * NP_ * D_;
  const _Float16* kp = k + (size_t)bh * NP_ * D_;
  const _Float16* vp = v + (size_t)bh * NP_ * D_;

  int qrow = tq * 16 + (lane & 15);
  if (qrow >= NP_) qrow = NP_ - 1;             // clamped (stores are guarded)
  const v16h qa0 = load_a_frag(qp + (size_t)qrow * D_, 0, lane);
  const v16h qa1 = load_a_frag(qp + (size_t)qrow * D_, 32, lane);

  float m[8], l[8];
  v8f o0 = {}, o1 = {}, o2 = {}, o3 = {};
#pragma unroll
  for (int r = 0; r < 8; ++r) { m[r] = -1e30f; l[r] = 0.f; }
  _Float16* pl = &p_lds[wave][0];

  // staging: chunk = 32 rows x 64 halves = 4KB = 256 x 16B segs; 128 threads
  // copy segs tid and tid+128 (flat, rows are contiguous).
  const unsigned lk0 = lds_off(&kbuf[0][tid * 8]);
  const unsigned lk1 = lds_off(&kbuf[1][tid * 8]);
  const unsigned lv0 = lds_off(&vbuf[0][tid * 8]);
  const unsigned lv1 = lds_off(&vbuf[1][tid * 8]);

  // preload chunk 0
  async_ld_b128(lk0,        kp + (size_t)tid * 8);
  async_ld_b128(lk0 + 2048, kp + (size_t)tid * 8 + 1024);
  async_ld_b128(lv0,        vp + (size_t)tid * 8);
  async_ld_b128(lv0 + 2048, vp + (size_t)tid * 8 + 1024);

  for (int i = 0; i < NP_ / 32; ++i) {         // 33 kv chunks
    const int kv = i * 32;
    wait_async0();
    __syncthreads();                           // chunk i resident for the block
    if (i + 1 < NP_ / 32) {
      const size_t src = (size_t)(kv + 32) * D_ + tid * 8;
      const unsigned dk = (i & 1) ? lk0 : lk1;
      const unsigned dv = (i & 1) ? lv0 : lv1;
      async_ld_b128(dk,        kp + src);
      async_ld_b128(dk + 2048, kp + src + 1024);
      async_ld_b128(dv,        vp + src);
      async_ld_b128(dv + 2048, vp + src + 1024);
    }
    const _Float16* kt = &kbuf[i & 1][0];      // [kvrow][d], ld=64
    const _Float16* vt = &vbuf[i & 1][0];

    v8f s0 = {}, s1 = {};
    s0 = wmma_f16(qa0, load_bt_frag(kt, D_, 0, 0, lane), s0);
    s0 = wmma_f16(qa1, load_bt_frag(kt, D_, 0, 32, lane), s0);
    s1 = wmma_f16(qa0, load_bt_frag(kt, D_, 16, 0, lane), s1);
    s1 = wmma_f16(qa1, load_bt_frag(kt, D_, 16, 32, lane), s1);

    const int c0 = kv + (lane & 15);           // kv column of this lane
    const int c1 = c0 + 16;
#pragma unroll
    for (int r = 0; r < 8; ++r) {
      float a0 = (c0 < N_) ? s0[r] * SCALE_ : -1e30f;   // mask padded kv cols
      float a1 = (c1 < N_) ? s1[r] * SCALE_ : -1e30f;
      float t = fmaxf(a0, a1);                  // row max across 32 cols
      t = fmaxf(t, __shfl_xor(t, 1));
      t = fmaxf(t, __shfl_xor(t, 2));
      t = fmaxf(t, __shfl_xor(t, 4));
      t = fmaxf(t, __shfl_xor(t, 8));
      const float mn = fmaxf(m[r], t);
      const float fs = __expf(m[r] - mn);
      const float p0 = __expf(a0 - mn);
      const float p1 = __expf(a1 - mn);
      float ps = p0 + p1;                       // row sum across 32 cols
      ps += __shfl_xor(ps, 1);
      ps += __shfl_xor(ps, 2);
      ps += __shfl_xor(ps, 4);
      ps += __shfl_xor(ps, 8);
      l[r] = l[r] * fs + ps;
      m[r] = mn;
      o0[r] *= fs; o1[r] *= fs; o2[r] *= fs; o3[r] *= fs;
      const int prow = r + 8 * g;               // C-frag row -> LDS row-major
      pl[prow * 32 + (lane & 15)]      = (_Float16)p0;
      pl[prow * 32 + 16 + (lane & 15)] = (_Float16)p1;
    }
    asm volatile("s_wait_dscnt 0" ::: "memory"); // per-wave P store->load

    const v16h pa = load_a_frag(pl + (size_t)(lane & 15) * 32, 0, lane);
    o0 = wmma_f16(pa, load_b_frag(vt, D_, 0, 0,  lane), o0);
    o1 = wmma_f16(pa, load_b_frag(vt, D_, 0, 16, lane), o1);
    o2 = wmma_f16(pa, load_b_frag(vt, D_, 0, 32, lane), o2);
    o3 = wmma_f16(pa, load_b_frag(vt, D_, 0, 48, lane), o3);
  }

#pragma unroll
  for (int r = 0; r < 8; ++r) {
    const int n = tq * 16 + r + 8 * g;
    if (n < N_) {
      const float inv = 1.f / l[r];
      const size_t base = ((size_t)bb * N_ + n) * C_ + hh * D_ + (lane & 15);
      ao[base]      = (_Float16)(o0[r] * inv);
      ao[base + 16] = (_Float16)(o1[r] * inv);
      ao[base + 32] = (_Float16)(o2[r] * inv);
      ao[base + 48] = (_Float16)(o3[r] * inv);
    }
  }
}

// ---------------------------------------------------------------- CLS attention
// cls[b, j-1] = mean_h softmax_j(q[b,h,0,:] . k[b,h,j,:] * scale)[j], j=1..N-1

__global__ __launch_bounds__(256) void cls_kernel(const _Float16* __restrict__ q,
                                                  const _Float16* __restrict__ k,
                                                  float* __restrict__ cls) {
  __shared__ float s[N_];
  __shared__ float acc[N_ - 1];
  __shared__ float red[256];
  const int b = blockIdx.x;
  const int tid = threadIdx.x;
  for (int j = tid; j < N_ - 1; j += 256) acc[j] = 0.f;
  __syncthreads();
  for (int h = 0; h < H_; ++h) {
    const _Float16* qp = q + (size_t)(b * H_ + h) * NP_ * D_;   // row 0
    const _Float16* kp = k + (size_t)(b * H_ + h) * NP_ * D_;
    for (int j = tid; j < N_; j += 256) {
      float dsum = 0.f;
#pragma unroll 8
      for (int c = 0; c < D_; ++c) dsum += (float)qp[c] * (float)kp[(size_t)j * D_ + c];
      s[j] = dsum * SCALE_;
    }
    __syncthreads();
    float lm = -1e30f;
    for (int j = tid; j < N_; j += 256) lm = fmaxf(lm, s[j]);
    red[tid] = lm;
    __syncthreads();
    for (int off = 128; off > 0; off >>= 1) {
      if (tid < off) red[tid] = fmaxf(red[tid], red[tid + off]);
      __syncthreads();
    }
    const float mx = red[0];
    __syncthreads();
    float ls = 0.f;
    for (int j = tid; j < N_; j += 256) { float e = __expf(s[j] - mx); s[j] = e; ls += e; }
    red[tid] = ls;
    __syncthreads();
    for (int off = 128; off > 0; off >>= 1) {
      if (tid < off) red[tid] += red[tid + off];
      __syncthreads();
    }
    const float inv = 1.f / (red[0] * (float)H_);
    __syncthreads();
    for (int j = tid; j < N_ - 1; j += 256) acc[j] += s[j + 1] * inv;
    __syncthreads();
  }
  for (int j = tid; j < N_ - 1; j += 256) cls[(size_t)b * (N_ - 1) + j] = acc[j];
}

// ---------------------------------------------------------------- proj GEMM
// Same async-staged tiling as qkv_gemm; f32 output + bias.

__global__ __launch_bounds__(256) void proj_gemm_kernel(
    const _Float16* __restrict__ ah, const _Float16* __restrict__ wph,
    const float* __restrict__ bias, float* __restrict__ out) {
  __shared__ _Float16 btile[2][64 * 32];
  const int tid = threadIdx.x;
  const int wave = tid >> 5;
  const int lane = tid & 31;
  const int g = (lane >> 4) & 1;
  const int NTC = C_ / 64;                    // 12 column groups
  const int mb = blockIdx.x / NTC;
  const int nt = blockIdx.x - mb * NTC;
  const int colbase = nt * 64;

  const int arow = mb * 128 + wave * 16 + (lane & 15);
  const _Float16* ap = ah + (size_t)(arow < M_ ? arow : (M_ - 1)) * C_;

  const int scol = tid >> 2, sseg = tid & 3;
  const _Float16* wsrc = wph + (size_t)(colbase + scol) * C_ + sseg * 8;
  const unsigned ldst0 = lds_off(&btile[0][scol * 32 + sseg * 8]);
  const unsigned ldst1 = lds_off(&btile[1][scol * 32 + sseg * 8]);

  async_ld_b128(ldst0, wsrc);
  v8f acc[4] = {};
  for (int c = 0; c < C_ / 32; ++c) {
    wait_async0();
    __syncthreads();
    if (c + 1 < C_ / 32)
      async_ld_b128((c & 1) ? ldst0 : ldst1, wsrc + (c + 1) * 32);
    v16h a = load_a_frag(ap, c * 32, lane);
    const _Float16* bt = &btile[c & 1][0];
#pragma unroll
    for (int t = 0; t < 4; ++t)
      acc[t] = wmma_f16(a, load_bt_frag(bt, 32, t * 16, 0, lane), acc[t]);
  }

#pragma unroll
  for (int t = 0; t < 4; ++t) {
    const int col = colbase + t * 16 + (lane & 15);
    const float bv = bias[col];
#pragma unroll
    for (int r = 0; r < 8; ++r) {
      const int orow = mb * 128 + wave * 16 + r + 8 * g;
      if (orow < M_) out[(size_t)orow * C_ + col] = acc[t][r] + bv;
    }
  }
}

// ---------------------------------------------------------------- launch

extern "C" void kernel_launch(void* const* d_in, const int* in_sizes, int n_in,
                              void* d_out, int out_size, void* d_ws, size_t ws_size,
                              hipStream_t stream) {
  (void)in_sizes; (void)n_in; (void)out_size; (void)ws_size;
  const float* x      = (const float*)d_in[0];
  const float* qkv_w  = (const float*)d_in[1];
  const float* proj_w = (const float*)d_in[2];
  const float* proj_b = (const float*)d_in[3];
  float* out = (float*)d_out;
  float* cls = out + (size_t)M_ * C_;

  char* ws = (char*)d_ws;
  size_t off = 0;
  auto alloc = [&](size_t bytes) -> char* {
    char* p = ws + off;
    off = (off + bytes + 255) & ~(size_t)255;
    return p;
  };
  _Float16* xh = (_Float16*)alloc((size_t)M_ * C_ * 2);
  _Float16* wq = (_Float16*)alloc((size_t)3 * C_ * C_ * 2);
  _Float16* wp = (_Float16*)alloc((size_t)C_ * C_ * 2);
  _Float16* qb = (_Float16*)alloc((size_t)B_ * H_ * NP_ * D_ * 2);
  _Float16* kb = (_Float16*)alloc((size_t)B_ * H_ * NP_ * D_ * 2);
  _Float16* vb = (_Float16*)alloc((size_t)B_ * H_ * NP_ * D_ * 2);
  _Float16* ao = (_Float16*)alloc((size_t)M_ * C_ * 2);

  // 1) f32 -> f16 conversions
  {
    int n = M_ * C_;
    f32_to_f16_kernel<<<(n + 255) / 256, 256, 0, stream>>>(x, xh, n);
    n = 3 * C_ * C_;
    f32_to_f16_kernel<<<(n + 255) / 256, 256, 0, stream>>>(qkv_w, wq, n);
    n = C_ * C_;
    f32_to_f16_kernel<<<(n + 255) / 256, 256, 0, stream>>>(proj_w, wp, n);
  }
  // 2) QKV projection: 65 row-blocks x 36 col-groups
  qkv_gemm_kernel<<<65 * 36, 256, 0, stream>>>(xh, wq, qb, kb, vb);
  // 3) flash attention: B*H blocks x 17 q-tile groups, 4 waves each
  attn_kernel<<<B_ * H_ * 17, 128, 0, stream>>>(qb, kb, vb, ao);
  // 4) CLS attention stats
  cls_kernel<<<B_, 256, 0, stream>>>(qb, kb, cls);
  // 5) output projection: 65 row-blocks x 12 col-groups
  proj_gemm_kernel<<<65 * 12, 256, 0, stream>>>(ao, wp, proj_b, out);
}
